// LSTM_69569880261298
// MI455X (gfx1250) — matrix operand order repaired and checked by reading, not compiled, not run on hardware
//
#include <hip/hip_runtime.h>
#include <hip/hip_bf16.h>

// ---------------------------------------------------------------------------
// LSTM (B=64, T=512, D=H=1024) for gfx1250 (CDNA5, wave32, WMMA).
//
//   Gx[t,b,4H] = x_t @ Wx.T + bias      (one big bf16 WMMA GEMM, parallel in T)
//   recurrent: single PERSISTENT kernel, 512 steps, grid-wide software
//              barrier per step; h double-buffered in packed-A bf16 layout,
//              per-block h tile staged in LDS via ASYNC loads, weights in L2.
//
// All matrix math: v_wmma_f32_16x16x32_bf16 (fp32 accumulate, fp32 cell/h).
// ---------------------------------------------------------------------------

#ifndef __has_builtin
#define __has_builtin(x) 0
#endif
#if __has_builtin(__builtin_amdgcn_global_load_async_to_lds_b128)
#define USE_ASYNC_LDS 1
#else
#define USE_ASYNC_LDS 0
#endif

#define LSTM_B 64
#define LSTM_T 512
#define LSTM_D 1024
#define LSTM_H 1024
#define NCOL   4096            // 4 gates * H
#define NT     256             // NCOL/16 column tiles
#define KT_D   32              // D/32  k-tiles
#define KT_H   32              // H/32  k-tiles
#define NBLK   256             // persistent grid: 4 bTiles * 64 jTiles

typedef __attribute__((ext_vector_type(16))) __bf16 v16bf;
typedef __attribute__((ext_vector_type(8)))  float  v8f;
typedef __attribute__((ext_vector_type(4)))  int    v4i;
typedef __attribute__((address_space(1))) v4i gv4i;   // global int4
typedef __attribute__((address_space(3))) v4i lv4i;   // LDS int4

union FragAB { uint4 q[2]; v16bf v; };   // 32 bytes = 16 bf16 per lane

__device__ __forceinline__ unsigned short f2bf(float f) {
    union { float f; unsigned u; } v; v.f = f;
    unsigned r = v.u + 0x7FFFu + ((v.u >> 16) & 1u);   // round-to-nearest-even
    return (unsigned short)(r >> 16);
}

__device__ __forceinline__ float sigf(float x) {
    return 1.0f / (1.0f + __expf(-x));
}

#if USE_ASYNC_LDS
__device__ __forceinline__ void wait_async0() {
#if __has_builtin(__builtin_amdgcn_s_wait_asynccnt)
    __builtin_amdgcn_s_wait_asynccnt(0);
#else
    asm volatile("s_wait_asynccnt 0x0" ::: "memory");
#endif
}
#endif

// generation-based grid-wide barrier (all NBLK blocks co-resident)
__device__ __forceinline__ void grid_sync(int* cnt, int* gen, int nb) {
    __syncthreads();
    if (threadIdx.x == 0) {
        int g = __hip_atomic_load(gen, __ATOMIC_RELAXED, __HIP_MEMORY_SCOPE_AGENT);
        __threadfence();                                   // release h/c writes
        int arrived = __hip_atomic_fetch_add(cnt, 1, __ATOMIC_ACQ_REL,
                                             __HIP_MEMORY_SCOPE_AGENT);
        if (arrived == nb - 1) {
            __hip_atomic_store(cnt, 0, __ATOMIC_RELAXED, __HIP_MEMORY_SCOPE_AGENT);
            __hip_atomic_fetch_add(gen, 1, __ATOMIC_ACQ_REL,
                                   __HIP_MEMORY_SCOPE_AGENT);
        } else {
            while (__hip_atomic_load(gen, __ATOMIC_ACQUIRE,
                                     __HIP_MEMORY_SCOPE_AGENT) == g)
                __builtin_amdgcn_s_sleep(2);
        }
    }
    __syncthreads();
}

// --------------------------- prep: cast x to bf16 ---------------------------
__global__ void k_cast_x(const float* __restrict__ x, unsigned short* __restrict__ xb, long n) {
    long i = (long)blockIdx.x * blockDim.x + threadIdx.x;
    if (i < n) xb[i] = f2bf(x[i]);
}

// ------------------- prep: pack weights into fragment order -----------------
// Fragment (kTile, nTile): 32x16 bf16. Lane L holds N = nTile*16 + (L&15),
// K = kTile*32 + (L>=16 ? 16 : 0) + e, e=0..15. Buffer index = frag*512 + L*16 + e.
__global__ void k_pack_w(const float* __restrict__ Wi, const float* __restrict__ Wf,
                         const float* __restrict__ Wo, const float* __restrict__ Wc,
                         unsigned short* __restrict__ wxp, unsigned short* __restrict__ whp) {
    long idx = (long)blockIdx.x * blockDim.x + threadIdx.x;
    const long HALF = (long)KT_D * NT * 512;                  // 4,194,304
    if (idx >= 2 * HALF) return;
    bool xpart = idx < HALF;
    long r = xpart ? idx : idx - HALF;
    int e  = (int)(r & 15);
    int L  = (int)((r >> 4) & 31);
    long frag = r >> 9;
    int nTile = (int)(frag & (NT - 1));
    int kTile = (int)(frag >> 8);
    int k = kTile * 32 + ((L >> 4) << 4) + e;
    int n = nTile * 16 + (L & 15);
    int gate = n >> 10;
    int j = n & 1023;
    const float* W = (gate == 0) ? Wi : (gate == 1) ? Wf : (gate == 2) ? Wo : Wc;
    int col = xpart ? k : (LSTM_D + k);                       // W is [H][D+H]
    float v = W[(long)j * (LSTM_D + LSTM_H) + col];
    (xpart ? wxp : whp)[r] = f2bf(v);
}

// --------------- prep: zero h (both bufs), c, and barrier state -------------
__global__ void k_init_state(unsigned short* hp0, unsigned short* hp1,
                             float* cBuf, int* bar) {
    int i = blockIdx.x * blockDim.x + threadIdx.x;            // 65536 total
    hp0[i] = 0; hp1[i] = 0; cBuf[i] = 0.0f;
    if (i == 0) { bar[0] = 0; bar[1] = 0; }
}

// ---------------- phase 1: Gx = x @ Wx.T + bias  (bf16 WMMA) ----------------
// One wave per 16x16 output tile. M = B*T rows, N = 4H cols, K = D.
__global__ void __launch_bounds__(256)
k_phase1(const unsigned short* __restrict__ xb, const uint4* __restrict__ wxp,
         const float* __restrict__ bi, const float* __restrict__ bfv,
         const float* __restrict__ bo, const float* __restrict__ bc,
         float* __restrict__ Gx) {
    int wave = threadIdx.x >> 5;
    int lane = threadIdx.x & 31;
    long tile = (long)blockIdx.x * 8 + wave;        // < 2048*256
    int  nTile = (int)(tile & (NT - 1));
    long mTile = tile >> 8;                         // < 2048

    int m = (int)(mTile * 16) + (lane & 15);        // A row handled by this lane
    const unsigned short* arow = xb + (long)m * LSTM_D;
    int aoff = (lane >> 4) * 16;                    // lanes 16-31: K chunk +16

    v8f acc = {};
#pragma unroll 4
    for (int kt = 0; kt < KT_D; ++kt) {
        int k0 = kt * 32;
        FragAB a, b;
        a.q[0] = *(const uint4*)(arow + k0 + aoff);        // K {0..7}/{16..23}
        a.q[1] = *(const uint4*)(arow + k0 + 8 + aoff);    // K {8..15}/{24..31}
        long fb = ((long)kt * NT + nTile) * 64 + lane * 2; // frag*64 uint4
        b.q[0] = wxp[fb];
        b.q[1] = wxp[fb + 1];
        acc = __builtin_amdgcn_wmma_f32_16x16x32_bf16(
            false, a.v, false, b.v, (short)0, acc, false, false);
    }

    int n = nTile * 16 + (lane & 15);
    int gate = n >> 10;
    int j = n & 1023;
    const float* bp = (gate == 0) ? bi : (gate == 1) ? bfv : (gate == 2) ? bo : bc;
    float bias = bp[j];

#pragma unroll
    for (int r = 0; r < 8; ++r) {
        int mr = (int)(mTile * 16) + r + ((lane >> 4) << 3);   // flat row = b*T + t
        int bb = mr >> 9;           // / T
        int tt = mr & 511;          // % T
        Gx[((long)(tt * LSTM_B + bb) * NCOL) + n] = acc[r] + bias;
    }
}

// ------------- persistent recurrent kernel: all 512 timesteps ---------------
// grid = 4 bTiles * 64 jTiles = 256 blocks; block = 128 (4 waves = gates i,f,o,c)
__global__ void __launch_bounds__(128)
k_recurrent(const uint4* __restrict__ hp0q, const uint4* __restrict__ hp1q,
            unsigned short* __restrict__ hp0w, unsigned short* __restrict__ hp1w,
            const uint4* __restrict__ whp, const float* __restrict__ Gx,
            float* __restrict__ cBuf, float* __restrict__ out,
            int* __restrict__ bar) {
    __shared__ float gbuf[4][256];
    __shared__ unsigned short hsh[KT_H * 512];     // 32KB packed h tile

    const int wave = threadIdx.x >> 5;
    const int lane = threadIdx.x & 31;
    const int bTile = blockIdx.x >> 6;             // 0..3
    const int jTile = blockIdx.x & 63;             // 0..63
    const int nTileG = wave * 64 + jTile;          // column tile: gate = wave
    const int nb = (int)gridDim.x;

    const int ncol = nTileG * 16 + (lane & 15);

    for (int t = 0; t < LSTM_T; ++t) {
        const uint4*    hR = (t & 1) ? hp1q : hp0q;
        unsigned short* hW = (t & 1) ? hp0w : hp1w;

        // stage this bTile's packed h_{t-1} into LDS (2048 uint4 = 32KB)
        {
            const uint4* src = hR + (long)bTile * (KT_H * 64);
            uint4* dst = (uint4*)hsh;
#if USE_ASYNC_LDS
#pragma unroll
            for (int i = 0; i < 16; ++i) {
                int idx = threadIdx.x + i * 128;
                __builtin_amdgcn_global_load_async_to_lds_b128(
                    (gv4i*)(src + idx), (lv4i*)(dst + idx), 0, 0);
            }
            wait_async0();                         // ASYNCcnt == 0: LDS writes done
#else
#pragma unroll
            for (int i = 0; i < 16; ++i)
                dst[threadIdx.x + i * 128] = src[threadIdx.x + i * 128];
#endif
        }
        __syncthreads();

        v8f acc = {};
        const uint4* hq = (const uint4*)hsh;
#pragma unroll 4
        for (int kt = 0; kt < KT_H; ++kt) {
            FragAB a, b;
            a.q[0] = hq[kt * 64 + lane * 2];                    // LDS
            a.q[1] = hq[kt * 64 + lane * 2 + 1];
            long fb = ((long)kt * NT + nTileG) * 64 + lane * 2; // L2-resident
            b.q[0] = whp[fb];
            b.q[1] = whp[fb + 1];
            acc = __builtin_amdgcn_wmma_f32_16x16x32_bf16(
                false, a.v, false, b.v, (short)0, acc, false, false);
        }

        // pre-activation = recurrent + (x-part + bias); stash in LDS.
        // Also prefetch next step's Gx slice to hide HBM latency behind the
        // barrier + next GEMM (lowers to global_prefetch).
        const float* gx  = Gx + (long)t * LSTM_B * NCOL;
        const float* gxn = gx + (long)LSTM_B * NCOL;
        bool pf = (t + 1) < LSTM_T;
#pragma unroll
        for (int r = 0; r < 8; ++r) {
            int mloc = r + ((lane >> 4) << 3);
            int bb = bTile * 16 + mloc;
            gbuf[wave][mloc * 16 + (lane & 15)] = acc[r] + gx[(long)bb * NCOL + ncol];
            if (pf) __builtin_prefetch(&gxn[(long)bb * NCOL + ncol], 0, 0);
        }
        __syncthreads();

        // pointwise: 256 elements, 128 threads x 2
#pragma unroll
        for (int ii = 0; ii < 2; ++ii) {
            int idx = threadIdx.x * 2 + ii;
            int mloc = idx >> 4;
            int jl   = idx & 15;
            int bb   = bTile * 16 + mloc;
            int jcol = jTile * 16 + jl;

            float si = sigf(gbuf[0][idx]);
            float sf = sigf(gbuf[1][idx]);
            float so = sigf(gbuf[2][idx]);
            float tg = tanhf(gbuf[3][idx]);

            long cix = (long)bb * LSTM_H + jcol;
            float cn = sf * cBuf[cix] + si * tg;
            cBuf[cix] = cn;
            float h = so * tanhf(cn);

            out[((long)t * LSTM_B + bb) * LSTM_H + jcol] = h;

            // write h in packed-A fragment order for next step (double-buffered)
            int kTile = jcol >> 5, kin = jcol & 31;
            int laneP = mloc + ((kin >= 16) ? 16 : 0);
            int e = kin & 15;
            hW[((long)bTile * KT_H + kTile) * 512 + laneP * 16 + e] = f2bf(h);
        }

        grid_sync(bar, bar + 1, nb);   // step t fully visible before t+1 reads
    }
}

// ---------------------------------------------------------------------------
extern "C" void kernel_launch(void* const* d_in, const int* in_sizes, int n_in,
                              void* d_out, int out_size, void* d_ws, size_t ws_size,
                              hipStream_t stream) {
    const float* x  = (const float*)d_in[0];
    const float* Wi = (const float*)d_in[1];
    const float* bi = (const float*)d_in[2];
    const float* Wf = (const float*)d_in[3];
    const float* bf = (const float*)d_in[4];
    const float* Wo = (const float*)d_in[5];
    const float* bo = (const float*)d_in[6];
    const float* Wc = (const float*)d_in[7];
    const float* bc = (const float*)d_in[8];
    float* out = (float*)d_out;

    // workspace layout
    char* ws = (char*)d_ws;
    size_t oXb  = 0;                                       // x bf16: 64MB
    size_t oWx  = oXb  + (size_t)LSTM_B * LSTM_T * LSTM_D * 2;
    size_t oWh  = oWx  + (size_t)KT_D * NT * 512 * 2;      // 8MB
    size_t oHp0 = oWh  + (size_t)KT_H * NT * 512 * 2;      // 8MB
    size_t oHp1 = oHp0 + (size_t)4 * KT_H * 512 * 2;       // 128KB
    size_t oC   = oHp1 + (size_t)4 * KT_H * 512 * 2;       // 128KB
    size_t oBar = oC   + (size_t)LSTM_B * LSTM_H * 4;      // 256KB
    size_t oGx  = oBar + 256;
    // Gx: T*B*4H fp32 = 512MB

    unsigned short* xb  = (unsigned short*)(ws + oXb);
    unsigned short* wxp = (unsigned short*)(ws + oWx);
    unsigned short* whp = (unsigned short*)(ws + oWh);
    unsigned short* hp0 = (unsigned short*)(ws + oHp0);
    unsigned short* hp1 = (unsigned short*)(ws + oHp1);
    float*          cB  = (float*)(ws + oC);
    int*            bar = (int*)(ws + oBar);
    float*          Gx  = (float*)(ws + oGx);

    // 1) cast x to bf16
    long nx = (long)LSTM_B * LSTM_T * LSTM_D;
    k_cast_x<<<(unsigned)((nx + 255) / 256), 256, 0, stream>>>(x, xb, nx);

    // 2) pack Wx and Wh fragments (bf16)
    long np = 2L * KT_D * NT * 512;
    k_pack_w<<<(unsigned)((np + 255) / 256), 256, 0, stream>>>(Wi, Wf, Wo, Wc, wxp, whp);

    // 3) zero h (both buffers), c, barrier
    k_init_state<<<65536 / 256, 256, 0, stream>>>(hp0, hp1, cB, bar);

    // 4) Gx = x @ Wx.T + bias for all T (bf16 WMMA GEMM)
    k_phase1<<<65536, 256, 0, stream>>>(xb, (const uint4*)wxp, bi, bf, bo, bc, Gx);

    // 5) single persistent kernel runs all 512 recurrent steps
    k_recurrent<<<NBLK, 128, 0, stream>>>((const uint4*)hp0, (const uint4*)hp1,
                                          hp0, hp1, (const uint4*)whp,
                                          Gx, cB, out, bar);
}
